// NeuromorphicSparseAttention_9062380994717
// MI455X (gfx1250) — compile-verified, matrix operand order checked by reading, not run on hardware
//
#include <hip/hip_runtime.h>
#include <hip/hip_bf16.h>

typedef __attribute__((ext_vector_type(16))) _Float16 v16h;
typedef __attribute__((ext_vector_type(8)))  _Float16 v8h;
typedef __attribute__((ext_vector_type(8)))  float    v8f;

#define B_   2
#define S_   2048
#define D_   1024
#define H_   16
#define HD_  64
#define NTOK (B_*S_)      /* 4096 */
#define QKV3 (3*D_)       /* 3072 */
#define NEH  (B_*S_*H_)   /* 65536 energy entries */

// ---------------------------------------------------------------------------
// helpers
// ---------------------------------------------------------------------------
__device__ __forceinline__ v16h mk16(v8h lo, v8h hi) {
  v16h r;
#pragma unroll
  for (int i = 0; i < 8; ++i) { r[i] = lo[i]; r[i + 8] = hi[i]; }
  return r;
}

// NT fragment loader for row-major [row][K] f16 matrices.
// 16-bit operand layout (ISA 7.12.2): lanes 0-15 hold K = kb+0..7 then kb+16..23,
// lanes 16-31 hold K = kb+8..15 then kb+24..31; row index = lane % 16.
__device__ __forceinline__ v16h load_frag_rm(const _Float16* base, int rowbase,
                                             int ld, int kb, int lane) {
  const int half = (lane >> 4) & 1;
  const int r    = lane & 15;
  const _Float16* p = base + (size_t)(rowbase + r) * ld + kb + half * 8;
  v8h lo = *(const v8h*)p;
  v8h hi = *(const v8h*)(p + 16);
  return mk16(lo, hi);
}

__device__ __forceinline__ v8f wmma16(v16h a, v16h b, v8f c) {
  return __builtin_amdgcn_wmma_f32_16x16x32_f16(false, a, false, b,
                                                (short)0, c, false, false);
}

// reductions over the 16-lane half this lane belongs to (xor masks < 16)
__device__ __forceinline__ float redmax16(float v) {
#pragma unroll
  for (int m = 8; m >= 1; m >>= 1) v = fmaxf(v, __shfl_xor(v, m, 32));
  return v;
}
__device__ __forceinline__ float redsum16(float v) {
#pragma unroll
  for (int m = 8; m >= 1; m >>= 1) v += __shfl_xor(v, m, 32);
  return v;
}

// ---------------------------------------------------------------------------
// kernel 1: fp32 -> f16 conversion
// ---------------------------------------------------------------------------
__global__ void f32_to_f16_kernel(const float* __restrict__ src,
                                  _Float16* __restrict__ dst, int n) {
  int i = blockIdx.x * 256 + threadIdx.x;
  if (i < n) dst[i] = (_Float16)src[i];
}

// ---------------------------------------------------------------------------
// kernel 2: QKV GEMM  (x[4096x1024] @ w_qkv^T[1024x3072])
// one wave computes a 32x64 tile (two A frags share every B frag).
// q,k stored per-head row-major [b][h][s][d]; v stored per-head TRANSPOSED
// [b][h][d][s] so attention's context-GEMM B fragments are contiguous loads.
// ---------------------------------------------------------------------------
__global__ __launch_bounds__(128) void qkv_gemm_kernel(
    const _Float16* __restrict__ xh, const _Float16* __restrict__ wqkvh,
    const float* __restrict__ b_qkv, _Float16* __restrict__ qh,
    _Float16* __restrict__ kh, _Float16* __restrict__ vt) {
  const int lane  = threadIdx.x & 31;
  const int w     = blockIdx.x * 4 + (threadIdx.x >> 5);  // 0..6143
  const int mtile = w % (NTOK / 32);                       // 0..127
  const int ntile = w / (NTOK / 32);                       // 0..47
  const int mbase = mtile * 32;
  const int nbase = ntile * 64;
  const int half  = lane >> 4;
  const int col   = lane & 15;

  v8f acc0[4], acc1[4];
#pragma unroll
  for (int i = 0; i < 4; ++i) { acc0[i] = (v8f)(0.0f); acc1[i] = (v8f)(0.0f); }

  for (int kb = 0; kb < D_; kb += 32) {
    v16h a0 = load_frag_rm(xh, mbase,      D_, kb, lane);
    v16h a1 = load_frag_rm(xh, mbase + 16, D_, kb, lane);
#pragma unroll
    for (int nt = 0; nt < 4; ++nt) {
      v16h bf = load_frag_rm(wqkvh, nbase + nt * 16, D_, kb, lane);
      acc0[nt] = wmma16(a0, bf, acc0[nt]);
      acc1[nt] = wmma16(a1, bf, acc1[nt]);
    }
  }

  // destination decode: nbase is 64-aligned -> single (matrix, head)
  const int mat = nbase >> 10;            // 0=q 1=k 2=v
  const int hh  = (nbase & 1023) >> 6;    // head
  const int bb  = mbase >> 11;            // batch (32-row tile never straddles)
  const int s0  = mbase & 2047;

#pragma unroll
  for (int nt = 0; nt < 4; ++nt) {
    const int dloc  = nt * 16 + col;
    const float bia = b_qkv[nbase + dloc];
#pragma unroll
    for (int g = 0; g < 2; ++g) {
#pragma unroll
      for (int r = 0; r < 8; ++r) {
        const int s = s0 + g * 16 + r + 8 * half;
        const float v = (g ? acc1[nt][r] : acc0[nt][r]) + bia;
        if (mat == 0) {
          qh[(((size_t)(bb * H_ + hh) * S_) + s) * HD_ + dloc] = (_Float16)v;
        } else if (mat == 1) {
          kh[(((size_t)(bb * H_ + hh) * S_) + s) * HD_ + dloc] = (_Float16)v;
        } else {
          vt[(((size_t)(bb * H_ + hh) * HD_) + dloc) * S_ + s] = (_Float16)v;
        }
      }
    }
  }
}

// ---------------------------------------------------------------------------
// kernel 3: per-(b,s,h) query energy + global sum for the sparsity threshold
// ---------------------------------------------------------------------------
__global__ __launch_bounds__(256) void energy_kernel(
    const _Float16* __restrict__ qh, float* __restrict__ energy,
    float* __restrict__ esum) {
  const int tid = blockIdx.x * 256 + threadIdx.x;  // 0..65535, = (b*S+s)*H+h
  const int b = tid >> 15;
  const int s = (tid >> 4) & (S_ - 1);
  const int h = tid & (H_ - 1);
  const _Float16* p = qh + (((size_t)(b * H_ + h) * S_) + s) * HD_;
  float a = 0.0f;
#pragma unroll
  for (int i = 0; i < HD_; ++i) {
    float v = (float)p[i];
    a += v * v;
  }
  const float e = sqrtf(a);
  energy[tid] = e;

  __shared__ float red[256];
  red[threadIdx.x] = e;
  __syncthreads();
  for (int st = 128; st > 0; st >>= 1) {
    if (threadIdx.x < st) red[threadIdx.x] += red[threadIdx.x + st];
    __syncthreads();
  }
  if (threadIdx.x == 0) atomicAdd(esum, red[0]);
}

// ---------------------------------------------------------------------------
// kernel 4: flash attention. 4 waves / block, 64 query rows per block (one
// 16-row tile per wave); waves stream the same K / V^T j-tiles in lockstep so
// they share them through the WGP cache. 64 columns per online-softmax step.
// ---------------------------------------------------------------------------
__global__ __launch_bounds__(128) void attn_kernel(
    const _Float16* __restrict__ qh, const _Float16* __restrict__ kh,
    const _Float16* __restrict__ vt, const float* __restrict__ energy,
    const float* __restrict__ esum, _Float16* __restrict__ ctxh) {
  __shared__ _Float16 sProbs[4][16 * 64];   // per-wave probs staging

  const int wave  = threadIdx.x >> 5;
  const int lane  = threadIdx.x & 31;
  const int half  = lane >> 4;
  const int col   = lane & 15;
  const int id    = blockIdx.x;          // 0..1023 = B*H*(S/64)
  const int b     = id >> 9;
  const int rem   = id & 511;
  const int h     = rem >> 5;
  const int it    = rem & 31;
  const int ibase = it * 64 + wave * 16;

  const size_t headoff = (size_t)(b * H_ + h) * S_ * HD_;
  const _Float16* qhead  = qh + headoff;
  const _Float16* khead  = kh + headoff;
  const _Float16* vthead = vt + headoff;  // [d][s], ld = S_

  const float thresh = 0.1f * (*esum) * (1.0f / (float)NEH);

  // resident Q fragments (K = 0..31 and 32..63)
  v16h qf[2];
#pragma unroll
  for (int t = 0; t < 2; ++t) qf[t] = load_frag_rm(qhead, ibase, HD_, t * 32, lane);

  // per-query-row mask*scale (1/sqrt(64) folded in); C-frag row map M = r + 8*half
  float mscale[8];
#pragma unroll
  for (int r = 0; r < 8; ++r) {
    const int m = r + 8 * half;
    const float e = energy[((size_t)(b * S_ + ibase + m)) * H_ + h];
    mscale[r] = (e > thresh) ? 0.125f : 0.0f;
  }

  float mrun[8], lsum[8];
  v8f acc[4];
#pragma unroll
  for (int r = 0; r < 8; ++r) { mrun[r] = -INFINITY; lsum[r] = 0.0f; }
#pragma unroll
  for (int d = 0; d < 4; ++d) acc[d] = (v8f)(0.0f);

  for (int jb = 0; jb < S_; jb += 64) {
    // prefetch next j-tile of K and V^T into cache (global_prefetch_b8)
    if (jb + 64 < S_) {
      __builtin_prefetch(khead + (size_t)(jb + 64 + lane) * HD_, 0, 1);
      __builtin_prefetch(vthead + (size_t)(lane * 2) * S_ + jb + 64, 0, 1);
    }

    // scores: four 16-col tiles, each accumulated over K = 64 (2 WMMAs)
    v8f c[4];
#pragma unroll
    for (int t = 0; t < 4; ++t) c[t] = (v8f)(0.0f);
#pragma unroll
    for (int t = 0; t < 4; ++t) {
#pragma unroll
      for (int u = 0; u < 2; ++u) {
        v16h kf = load_frag_rm(khead, jb + t * 16, HD_, u * 32, lane);
        c[t] = wmma16(qf[u], kf, c[t]);
      }
    }

    // online softmax update + stage probs (f16) into LDS in row-major A order
#pragma unroll
    for (int r = 0; r < 8; ++r) {
      float x0 = c[0][r] * mscale[r];
      float x1 = c[1][r] * mscale[r];
      float x2 = c[2][r] * mscale[r];
      float x3 = c[3][r] * mscale[r];
      const float mx   = redmax16(fmaxf(fmaxf(x0, x1), fmaxf(x2, x3)));
      const float mnew = fmaxf(mrun[r], mx);
      const float al   = __expf(mrun[r] - mnew);
      const float p0 = __expf(x0 - mnew);
      const float p1 = __expf(x1 - mnew);
      const float p2 = __expf(x2 - mnew);
      const float p3 = __expf(x3 - mnew);
      lsum[r] = lsum[r] * al + redsum16((p0 + p1) + (p2 + p3));
      mrun[r] = mnew;
#pragma unroll
      for (int d = 0; d < 4; ++d) acc[d][r] *= al;
      const int m = r + 8 * half;
      sProbs[wave][m * 64 + col]      = (_Float16)p0;
      sProbs[wave][m * 64 + 16 + col] = (_Float16)p1;
      sProbs[wave][m * 64 + 32 + col] = (_Float16)p2;
      sProbs[wave][m * 64 + 48 + col] = (_Float16)p3;
    }
    __syncthreads();   // make probs stores visible across lanes

    // context: acc += probs(16x64) @ v(64x64); V^T fragments straight from global
    v16h pf[2];
#pragma unroll
    for (int t = 0; t < 2; ++t) {
      const _Float16* p = &sProbs[wave][col * 64 + t * 32 + half * 8];
      pf[t] = mk16(*(const v8h*)p, *(const v8h*)(p + 16));
    }
#pragma unroll
    for (int dt = 0; dt < 4; ++dt) {
#pragma unroll
      for (int t = 0; t < 2; ++t) {
        v16h vf = load_frag_rm(vthead, dt * 16, S_, jb + t * 32, lane);
        acc[dt] = wmma16(pf[t], vf, acc[dt]);
      }
    }
    __syncthreads();   // probs buffer reusable next iteration
  }

  // normalize and store context as (B,S,D) f16, D index = h*64 + d
  const size_t outbase = ((size_t)b * S_ + ibase) * (size_t)D_ + h * HD_;
#pragma unroll
  for (int dt = 0; dt < 4; ++dt) {
#pragma unroll
    for (int r = 0; r < 8; ++r) {
      const int m = r + 8 * half;
      ctxh[outbase + (size_t)m * D_ + dt * 16 + col] =
          (_Float16)(acc[dt][r] / lsum[r]);
    }
  }
}

// ---------------------------------------------------------------------------
// kernel 5: output projection (ctx[4096x1024] @ w_out^T[1024x1024]) + bias -> f32
// one wave computes a 32x64 tile
// ---------------------------------------------------------------------------
__global__ __launch_bounds__(128) void out_gemm_kernel(
    const _Float16* __restrict__ ctxh, const _Float16* __restrict__ wouth,
    const float* __restrict__ b_out, float* __restrict__ out) {
  const int lane  = threadIdx.x & 31;
  const int w     = blockIdx.x * 4 + (threadIdx.x >> 5);  // 0..2047
  const int mtile = w % (NTOK / 32);                       // 0..127
  const int ntile = w / (NTOK / 32);                       // 0..15
  const int mbase = mtile * 32;
  const int nbase = ntile * 64;
  const int half  = lane >> 4;
  const int col   = lane & 15;

  v8f acc0[4], acc1[4];
#pragma unroll
  for (int i = 0; i < 4; ++i) { acc0[i] = (v8f)(0.0f); acc1[i] = (v8f)(0.0f); }

  for (int kb = 0; kb < D_; kb += 32) {
    v16h a0 = load_frag_rm(ctxh, mbase,      D_, kb, lane);
    v16h a1 = load_frag_rm(ctxh, mbase + 16, D_, kb, lane);
#pragma unroll
    for (int nt = 0; nt < 4; ++nt) {
      v16h bf = load_frag_rm(wouth, nbase + nt * 16, D_, kb, lane);
      acc0[nt] = wmma16(a0, bf, acc0[nt]);
      acc1[nt] = wmma16(a1, bf, acc1[nt]);
    }
  }

#pragma unroll
  for (int nt = 0; nt < 4; ++nt) {
    const int gcol = nbase + nt * 16 + col;
    const float bia = b_out[gcol];
#pragma unroll
    for (int g = 0; g < 2; ++g) {
#pragma unroll
      for (int r = 0; r < 8; ++r) {
        const int row = mbase + g * 16 + r + 8 * half;
        out[(size_t)row * D_ + gcol] = (g ? acc1[nt][r] : acc0[nt][r]) + bia;
      }
    }
  }
}

// ---------------------------------------------------------------------------
// launch
// ---------------------------------------------------------------------------
extern "C" void kernel_launch(void* const* d_in, const int* in_sizes, int n_in,
                              void* d_out, int out_size, void* d_ws, size_t ws_size,
                              hipStream_t stream) {
  const float* x     = (const float*)d_in[0];
  const float* w_qkv = (const float*)d_in[1];
  const float* b_qkv = (const float*)d_in[2];
  const float* w_out = (const float*)d_in[3];
  const float* b_out = (const float*)d_in[4];

  char* ws = (char*)d_ws;
  size_t off = 0;
  _Float16* xh    = (_Float16*)(ws + off); off += (size_t)NTOK * D_ * 2;   // 8 MB
  _Float16* wqkvh = (_Float16*)(ws + off); off += (size_t)QKV3 * D_ * 2;   // 6 MB
  _Float16* wouth = (_Float16*)(ws + off); off += (size_t)D_ * D_ * 2;     // 2 MB
  _Float16* qh    = (_Float16*)(ws + off); off += (size_t)NTOK * D_ * 2;   // 8 MB
  _Float16* kh    = (_Float16*)(ws + off); off += (size_t)NTOK * D_ * 2;   // 8 MB
  _Float16* vt    = (_Float16*)(ws + off); off += (size_t)NTOK * D_ * 2;   // 8 MB
  _Float16* ctxh  = (_Float16*)(ws + off); off += (size_t)NTOK * D_ * 2;   // 8 MB
  float*    energy= (float*)(ws + off);    off += (size_t)NEH * 4;         // 256 KB
  float*    esum  = (float*)(ws + off);    off += 256;

  f32_to_f16_kernel<<<(NTOK * D_ + 255) / 256, 256, 0, stream>>>(x, xh, NTOK * D_);
  f32_to_f16_kernel<<<(QKV3 * D_ + 255) / 256, 256, 0, stream>>>(w_qkv, wqkvh, QKV3 * D_);
  f32_to_f16_kernel<<<(D_ * D_ + 255) / 256, 256, 0, stream>>>(w_out, wouth, D_ * D_);

  // 6144 waves (128 m-tiles x 48 n-tiles), 4 waves/block
  qkv_gemm_kernel<<<1536, 128, 0, stream>>>(xh, wqkvh, b_qkv, qh, kh, vt);

  hipMemsetAsync(esum, 0, sizeof(float), stream);
  energy_kernel<<<NEH / 256, 256, 0, stream>>>(qh, energy, esum);

  // one block per (b, h, 64-row i-tile): 2*16*32 = 1024 blocks, 4 waves each
  attn_kernel<<<1024, 128, 0, stream>>>(qh, kh, vt, energy, esum, ctxh);

  // 2048 waves (128 m-tiles x 16 n-tiles), 4 waves/block
  out_gemm_kernel<<<512, 128, 0, stream>>>(ctxh, wouth, b_out, (float*)d_out);
}
